// ED_Mamba_20942260535426
// MI455X (gfx1250) — matrix operand order repaired and checked by reading, not compiled, not run on hardware
//
#include <hip/hip_runtime.h>
#include <hip/hip_bf16.h>
#include <math.h>

typedef __attribute__((ext_vector_type(16))) _Float16 h16;
typedef __attribute__((ext_vector_type(8)))  _Float16 h8;
typedef __attribute__((ext_vector_type(2)))  _Float16 h2;
typedef __attribute__((ext_vector_type(8)))  float    f8;

#define HH 256
#define WWID 256
#define NBATCH 8
#define NPIX (NBATCH*HH*WWID)   // 524288 pixels
#define CCH 32

__device__ __forceinline__ float sigmoidf(float v) { return 1.0f / (1.0f + __expf(-v)); }
__device__ __forceinline__ float lrelu(float v) { return v > 0.0f ? v : 0.01f * v; }

// ---------------------------------------------------------------------------
// Weight packing: OIHW fp32 -> WMMA A-fragment-ordered fp16.
// Fragment f = ((chtile*NTAPS + tap)*NKB + s); within fragment, lane l holds
// row M = l%16, elements e hold K = (l>=16?16:0) + e.
// Input channel index i = s*32 + K; zero-fill when i >= Cin (f4's 33-ch case).
// ---------------------------------------------------------------------------
__global__ void pack_weights(const float* __restrict__ w, int Cin, int ntaps,
                             int nkb, _Float16* __restrict__ out) {
  int idx = blockIdx.x * 256 + threadIdx.x;
  int total = 2 * ntaps * nkb * 32 * 16;
  if (idx >= total) return;
  int e    = idx & 15;
  int lane = (idx >> 4) & 31;
  int f    = idx >> 9;
  int s      = f % nkb;
  int tap    = (f / nkb) % ntaps;
  int chtile = f / (nkb * ntaps);
  int o = chtile * 16 + (lane & 15);
  int K = ((lane >> 4) * 16) + e;
  int i = s * 32 + K;
  float v = 0.0f;
  if (i < Cin) v = w[((size_t)o * Cin + i) * ntaps + tap];
  out[idx] = (_Float16)v;
}

// ---------------------------------------------------------------------------
// Sobel edge conv (1->32ch, 3x3, pad 1) fused with concat([x, out0]) staging
// into a zero-padded 64-channel NHWC fp16 tensor (ch0=x, ch1..32=sobel, rest 0).
// ---------------------------------------------------------------------------
__global__ __launch_bounds__(256)
void sobel_stage(const float* __restrict__ x, const float* __restrict__ base,
                 const float* __restrict__ factor, const float* __restrict__ bias,
                 _Float16* __restrict__ stage) {
  long pix = (long)blockIdx.x * 256 + threadIdx.x;
  int b = (int)(pix >> 16);
  int h = (int)((pix >> 8) & 255);
  int w = (int)(pix & 255);
  const float* xb = x + ((long)b << 16);
  float p[9];
#pragma unroll
  for (int ky = 0; ky < 3; ++ky)
#pragma unroll
    for (int kx = 0; kx < 3; ++kx) {
      int hh = h + ky - 1, ww = w + kx - 1;
      bool ok = (hh >= 0) && (hh < HH) && (ww >= 0) && (ww < WWID);
      p[ky * 3 + kx] = ok ? xb[hh * WWID + ww] : 0.0f;
    }
  h16 v0 = {}, v1 = {}, v2 = {}, v3 = {};
  v0[0] = (_Float16)p[4];  // center x value (channel 0 of concat)
#pragma unroll
  for (int c = 0; c < 32; ++c) {
    float g = 0.0f;
#pragma unroll
    for (int t = 0; t < 9; ++t) g += base[c * 9 + t] * p[t];
    float val = g * factor[c] + bias[c];
    if (c < 15)      v0[c + 1]  = (_Float16)val;
    else if (c < 31) v1[c - 15] = (_Float16)val;
    else             v2[0]      = (_Float16)val;
  }
  _Float16* d = stage + pix * 64;
  *(h16*)(d)      = v0;
  *(h16*)(d + 16) = v1;
  *(h16*)(d + 32) = v2;
  *(h16*)(d + 48) = v3;
}

// ---------------------------------------------------------------------------
// Implicit-GEMM conv via WMMA. Block = 256 threads (8 waves), tile =
// 128 pixels x 32 out-channels. Each wave owns 16 pixels and computes BOTH
// 16-channel output tiles so every B-fragment load feeds two WMMAs.
// K-block s reads 32 consecutive channels of srcs[s]; NTAPS=9 (3x3, dilated)
// or 1 (1x1). ACT: 0=leaky_relu(0.01), 1=relu. Output NHWC fp16, C=32.
// A-fragments are loaded inside the tap loop (used once; avoids pinning
// 18 fragments in VGPRs and killing occupancy).
// ---------------------------------------------------------------------------
template <int NTAPS, int NKB, int ACT>
__global__ __launch_bounds__(256)
void conv_wmma(const _Float16* __restrict__ src0, int c0,
               const _Float16* __restrict__ src1, int c1,
               const _Float16* __restrict__ wpack,
               const float* __restrict__ bias,
               _Float16* __restrict__ dst, int dil) {
  int lane = threadIdx.x & 31;
  int wv   = threadIdx.x >> 5;
  long pix0 = (long)blockIdx.x * 128;
  int b  = (int)(pix0 >> 16);
  int h  = (int)((pix0 >> 8) & 255);
  int w0 = (int)(pix0 & 255) + wv * 16;
  int n  = lane & 15;
  int lh = lane >> 4;

  f8 acc0 = {};  // out-channels 0..15
  f8 acc1 = {};  // out-channels 16..31
#pragma unroll
  for (int t = 0; t < NTAPS; ++t) {
    int dy = (NTAPS == 9) ? (t / 3 - 1) * dil : 0;
    int dx = (NTAPS == 9) ? (t % 3 - 1) * dil : 0;
    int hh = h + dy;
    int ww = w0 + n + dx;
    bool ok = (hh >= 0) & (hh < HH) & (ww >= 0) & (ww < WWID);
    long pix = ((long)b * HH + hh) * WWID + ww;
#pragma unroll
    for (int s = 0; s < NKB; ++s) {
      const _Float16* sp = (s == 0) ? src0 : src1;
      long cs = (s == 0) ? c0 : c1;
      h16 bfrag = {};
      if (ok) bfrag = *(const h16*)(sp + pix * cs + lh * 16);
      h16 a0 = *(const h16*)(wpack + ((size_t)((0 * NTAPS + t) * NKB + s) * 32 + lane) * 16);
      h16 a1 = *(const h16*)(wpack + ((size_t)((1 * NTAPS + t) * NKB + s) * 32 + lane) * 16);
      acc0 = __builtin_amdgcn_wmma_f32_16x16x32_f16(
          false, a0, false, bfrag, (short)0, acc0, false, false);
      acc1 = __builtin_amdgcn_wmma_f32_16x16x32_f16(
          false, a1, false, bfrag, (short)0, acc1, false, false);
    }
  }

  // Epilogue: bias + activation, pack 8 consecutive channels, 16B stores.
  long opix = ((long)b * HH + h) * WWID + w0 + n;
  int cb0 = lh * 8;        // from acc0
  int cb1 = 16 + lh * 8;   // from acc1
  h8 o0, o1;
#pragma unroll
  for (int r = 0; r < 8; ++r) {
    float v0 = acc0[r] + bias[cb0 + r];
    float v1 = acc1[r] + bias[cb1 + r];
    v0 = (ACT == 0) ? lrelu(v0) : fmaxf(v0, 0.0f);
    v1 = (ACT == 0) ? lrelu(v1) : fmaxf(v1, 0.0f);
    o0[r] = (_Float16)v0;
    o1[r] = (_Float16)v1;
  }
  *(h8*)(dst + opix * 32 + cb0) = o0;
  *(h8*)(dst + opix * 32 + cb1) = o1;
}

// ---------------------------------------------------------------------------
// Softmax gate over {ex,ey,ez}: 3x 96-dot per pixel, softmax, blend -> enh.
// ---------------------------------------------------------------------------
__global__ __launch_bounds__(256)
void gate_mix(const _Float16* __restrict__ ex, const _Float16* __restrict__ ey,
              const _Float16* __restrict__ ez, const float* __restrict__ gw,
              const float* __restrict__ gb, _Float16* __restrict__ enh) {
  long pix = (long)blockIdx.x * 256 + threadIdx.x;
  h16 x0 = *(const h16*)(ex + pix * 32);
  h16 x1 = *(const h16*)(ex + pix * 32 + 16);
  h16 y0 = *(const h16*)(ey + pix * 32);
  h16 y1 = *(const h16*)(ey + pix * 32 + 16);
  h16 z0 = *(const h16*)(ez + pix * 32);
  h16 z1 = *(const h16*)(ez + pix * 32 + 16);
  float g[3];
#pragma unroll
  for (int j = 0; j < 3; ++j) {
    float s = gb[j];
    const float* wj = gw + j * 96;
#pragma unroll
    for (int c = 0; c < 16; ++c) {
      s += wj[c] * (float)x0[c] + wj[16 + c] * (float)x1[c];
      s += wj[32 + c] * (float)y0[c] + wj[48 + c] * (float)y1[c];
      s += wj[64 + c] * (float)z0[c] + wj[80 + c] * (float)z1[c];
    }
    g[j] = s;
  }
  float m = fmaxf(g[0], fmaxf(g[1], g[2]));
  float e0 = __expf(g[0] - m), e1 = __expf(g[1] - m), e2 = __expf(g[2] - m);
  float inv = 1.0f / (e0 + e1 + e2);
  e0 *= inv; e1 *= inv; e2 *= inv;
  h16 o0, o1;
#pragma unroll
  for (int c = 0; c < 16; ++c) {
    o0[c] = (_Float16)(e0 * (float)x0[c] + e1 * (float)y0[c] + e2 * (float)z0[c]);
    o1[c] = (_Float16)(e0 * (float)x1[c] + e1 * (float)y1[c] + e2 * (float)z1[c]);
  }
  *(h16*)(enh + pix * 32) = o0;
  *(h16*)(enh + pix * 32 + 16) = o1;
}

// ---------------------------------------------------------------------------
// Global spatial avg/max pool per (b,c): one block per (b,c), LDS reduction.
// ---------------------------------------------------------------------------
__global__ __launch_bounds__(256)
void pool_bc(const _Float16* __restrict__ enh, float* __restrict__ avg,
             float* __restrict__ mx) {
  int bc = blockIdx.x;
  int b = bc >> 5, c = bc & 31;
  const _Float16* p = enh + ((long)b << 16) * 32 + c;
  float s = 0.0f, m = -3.4e38f;
  for (int i = threadIdx.x; i < HH * WWID; i += 256) {
    float v = (float)p[(long)i * 32];
    s += v;
    m = fmaxf(m, v);
  }
  __shared__ float ss[256], sm[256];
  ss[threadIdx.x] = s; sm[threadIdx.x] = m;
  __syncthreads();
  for (int st = 128; st > 0; st >>= 1) {
    if (threadIdx.x < st) {
      ss[threadIdx.x] += ss[threadIdx.x + st];
      sm[threadIdx.x] = fmaxf(sm[threadIdx.x], sm[threadIdx.x + st]);
    }
    __syncthreads();
  }
  if (threadIdx.x == 0) {
    avg[bc] = ss[0] * (1.0f / (HH * WWID));
    mx[bc] = sm[0];
  }
}

// Channel attention MLP (32->2->32, shared weights on avg & max paths).
__global__ void ca_fc(const float* __restrict__ avg, const float* __restrict__ mx,
                      const float* __restrict__ w1, const float* __restrict__ w2,
                      float* __restrict__ ca) {
  int t = threadIdx.x;  // 256 threads: b = t>>5, c = t&31
  int b = t >> 5, c = t & 31;
  float ha[2], hm[2];
#pragma unroll
  for (int j = 0; j < 2; ++j) {
    float sa = 0.0f, sm = 0.0f;
#pragma unroll
    for (int k = 0; k < 32; ++k) {
      sa += w1[j * 32 + k] * avg[b * 32 + k];
      sm += w1[j * 32 + k] * mx[b * 32 + k];
    }
    ha[j] = fmaxf(sa, 0.0f);
    hm[j] = fmaxf(sm, 0.0f);
  }
  float o = 0.0f;
#pragma unroll
  for (int j = 0; j < 2; ++j) o += w2[c * 2 + j] * (ha[j] + hm[j]);
  ca[t] = sigmoidf(o);
}

// f = ca*enh; also emit per-pixel channel mean/max -> sa_in (2ch fp16).
__global__ __launch_bounds__(256)
void apply_ca(const _Float16* __restrict__ enh, const float* __restrict__ ca,
              _Float16* __restrict__ f, _Float16* __restrict__ sain) {
  long pix = (long)blockIdx.x * 256 + threadIdx.x;
  int b = (int)(pix >> 16);
  h16 a0 = *(const h16*)(enh + pix * 32);
  h16 a1 = *(const h16*)(enh + pix * 32 + 16);
  const float* cab = ca + b * 32;
  h16 o0, o1;
  float sum = 0.0f, m = -3.4e38f;
#pragma unroll
  for (int c = 0; c < 16; ++c) {
    float v0 = cab[c] * (float)a0[c];
    float v1 = cab[16 + c] * (float)a1[c];
    o0[c] = (_Float16)v0; o1[c] = (_Float16)v1;
    sum += v0 + v1;
    m = fmaxf(m, fmaxf(v0, v1));
  }
  *(h16*)(f + pix * 32) = o0;
  *(h16*)(f + pix * 32 + 16) = o1;
  h2 s; s[0] = (_Float16)(sum * (1.0f / 32.0f)); s[1] = (_Float16)m;
  *(h2*)(sain + pix * 2) = s;
}

// Spatial attention 7x7 conv (2->1, pad 3) + sigmoid, then
// out0 = lrelu(out1 + sa * f).
__global__ __launch_bounds__(256)
void sa_merge(const _Float16* __restrict__ sain, const float* __restrict__ saw,
              const _Float16* __restrict__ f, const _Float16* __restrict__ out1,
              _Float16* __restrict__ out0) {
  long pix = (long)blockIdx.x * 256 + threadIdx.x;
  int b = (int)(pix >> 16);
  int h = (int)((pix >> 8) & 255);
  int w = (int)(pix & 255);
  float s = 0.0f;
#pragma unroll
  for (int ky = 0; ky < 7; ++ky) {
    int hh = h + ky - 3;
    if (hh < 0 || hh >= HH) continue;
#pragma unroll
    for (int kx = 0; kx < 7; ++kx) {
      int ww = w + kx - 3;
      if (ww < 0 || ww >= WWID) continue;
      long q = ((long)b * HH + hh) * WWID + ww;
      h2 v = *(const h2*)(sain + q * 2);
      s += saw[ky * 7 + kx] * (float)v[0] + saw[49 + ky * 7 + kx] * (float)v[1];
    }
  }
  float sa = sigmoidf(s);
  h16 f0 = *(const h16*)(f + pix * 32);
  h16 f1 = *(const h16*)(f + pix * 32 + 16);
  h16 u0 = *(const h16*)(out1 + pix * 32);
  h16 u1 = *(const h16*)(out1 + pix * 32 + 16);
  h16 o0, o1;
#pragma unroll
  for (int c = 0; c < 16; ++c) {
    o0[c] = (_Float16)lrelu((float)u0[c] + sa * (float)f0[c]);
    o1[c] = (_Float16)lrelu((float)u1[c] + sa * (float)f1[c]);
  }
  *(h16*)(out0 + pix * 32) = o0;
  *(h16*)(out0 + pix * 32 + 16) = o1;
}

// Final: concat(out0, t3) NHWC fp16 -> NCHW fp32 into d_out.
__global__ __launch_bounds__(256)
void final_out(const _Float16* __restrict__ out0, const _Float16* __restrict__ t3,
               float* __restrict__ out) {
  long idx = (long)blockIdx.x * 256 + threadIdx.x;  // 8*64*256*256 elements
  int w = (int)(idx & 255);
  int h = (int)((idx >> 8) & 255);
  int c = (int)((idx >> 16) & 63);
  int b = (int)(idx >> 22);
  long pix = ((long)b * HH + h) * WWID + w;
  const _Float16* src = (c < 32) ? out0 : t3;
  out[idx] = (float)src[pix * 32 + (c & 31)];
}

// ---------------------------------------------------------------------------
extern "C" void kernel_launch(void* const* d_in, const int* in_sizes, int n_in,
                              void* d_out, int out_size, void* d_ws, size_t ws_size,
                              hipStream_t stream) {
  const float* x       = (const float*)d_in[0];
  const float* sb_base = (const float*)d_in[1];
  const float* sb_fac  = (const float*)d_in[2];
  const float* sb_bias = (const float*)d_in[3];
  const float* f4_w = (const float*)d_in[4];  const float* f4_b = (const float*)d_in[5];
  const float* ec1_w = (const float*)d_in[6]; const float* ec1_b = (const float*)d_in[7];
  const float* ec2_w = (const float*)d_in[8]; const float* ec2_b = (const float*)d_in[9];
  const float* ec3_w = (const float*)d_in[10];const float* ec3_b = (const float*)d_in[11];
  const float* gate_w = (const float*)d_in[12]; const float* gate_b = (const float*)d_in[13];
  const float* ca1_w = (const float*)d_in[14];
  const float* ca2_w = (const float*)d_in[15];
  const float* sa_w  = (const float*)d_in[16];
  const float* p1_w = (const float*)d_in[17]; const float* p1_b = (const float*)d_in[18];
  const float* f1_w = (const float*)d_in[19]; const float* f1_b = (const float*)d_in[20];
  const float* p2_w = (const float*)d_in[21]; const float* p2_b = (const float*)d_in[22];
  const float* f2_w = (const float*)d_in[23]; const float* f2_b = (const float*)d_in[24];
  const float* p3_w = (const float*)d_in[25]; const float* p3_b = (const float*)d_in[26];
  const float* f3_w = (const float*)d_in[27]; const float* f3_b = (const float*)d_in[28];

  const size_t M32 = 33554432;  // 32MB region = NPIX * 32ch * 2B
  char* ws = (char*)d_ws;
  _Float16* R1   = (_Float16*)(ws);             // out1 -> t3
  _Float16* R2   = (_Float16*)(ws + M32);       // ez -> f -> t2
  _Float16* R3   = (_Float16*)(ws + 2 * M32);   // enh -> out0 (live to end)
  _Float16* SAIN = (_Float16*)(ws + 3 * M32);   // 2MB
  char* misc = ws + 3 * M32 + 2097152;
  float* AVG = (float*)misc;           // 256 floats
  float* MAXP = AVG + 256;
  float* CA = MAXP + 256;
  _Float16* WPK = (_Float16*)(misc + 4096);
  _Float16* wpk_f4  = WPK;                size_t off = 2 * 9 * 2 * 512;
  _Float16* wpk_ec1 = WPK + off;          off += 2 * 9 * 1 * 512;
  _Float16* wpk_ec2 = WPK + off;          off += 2 * 9 * 1 * 512;
  _Float16* wpk_ec3 = WPK + off;          off += 2 * 9 * 1 * 512;
  _Float16* wpk_p1  = WPK + off;          off += 2 * 1 * 1 * 512;
  _Float16* wpk_f1  = WPK + off;          off += 2 * 9 * 1 * 512;
  _Float16* wpk_p2  = WPK + off;          off += 2 * 1 * 2 * 512;
  _Float16* wpk_f2  = WPK + off;          off += 2 * 9 * 1 * 512;
  _Float16* wpk_p3  = WPK + off;          off += 2 * 1 * 2 * 512;
  _Float16* wpk_f3  = WPK + off;

  // d_out (128MB) doubles as scratch until the final conversion.
  char* ob = (char*)d_out;
  _Float16* STAGE = (_Float16*)ob;           // 64MB: concat([x,sobel,0]) NHWC C=64
  _Float16* EX = (_Float16*)ob;              // 32MB (after stage is dead)
  _Float16* EY = (_Float16*)(ob + M32);      // 32MB
  _Float16* ABUF = EX;                       // p1/p2/p3 outputs
  _Float16* TBUF = EY;                       // t1

  const int EW = 2048;                 // elementwise grid: NPIX/256
  const int CG = NPIX / 128;           // conv grid: 4096 (128 px per block)

  // Pack all conv weights into WMMA fragment order.
  pack_weights<<<72, 256, 0, stream>>>(f4_w, 33, 9, 2, wpk_f4);
  pack_weights<<<36, 256, 0, stream>>>(ec1_w, 32, 9, 1, wpk_ec1);
  pack_weights<<<36, 256, 0, stream>>>(ec2_w, 32, 9, 1, wpk_ec2);
  pack_weights<<<36, 256, 0, stream>>>(ec3_w, 32, 9, 1, wpk_ec3);
  pack_weights<<<4, 256, 0, stream>>>(p1_w, 32, 1, 1, wpk_p1);
  pack_weights<<<36, 256, 0, stream>>>(f1_w, 32, 9, 1, wpk_f1);
  pack_weights<<<8, 256, 0, stream>>>(p2_w, 64, 1, 2, wpk_p2);
  pack_weights<<<36, 256, 0, stream>>>(f2_w, 32, 9, 1, wpk_f2);
  pack_weights<<<8, 256, 0, stream>>>(p3_w, 64, 1, 2, wpk_p3);
  pack_weights<<<36, 256, 0, stream>>>(f3_w, 32, 9, 1, wpk_f3);

  // Sobel + concat staging (C=64, zero-padded).
  sobel_stage<<<EW, 256, 0, stream>>>(x, sb_base, sb_fac, sb_bias, STAGE);

  // out1 = lrelu(conv3x3(stage, f4)) : K-blocks = stage[0:32], stage[32:64]
  conv_wmma<9, 2, 0><<<CG, 256, 0, stream>>>(STAGE, 64, STAGE + 32, 64,
                                             wpk_f4, f4_b, R1, 1);
  // ex/ey/ez = relu(dilated 3x3 convs)
  conv_wmma<9, 1, 1><<<CG, 256, 0, stream>>>(R1, 32, R1, 32, wpk_ec1, ec1_b, EX, 1);
  conv_wmma<9, 1, 1><<<CG, 256, 0, stream>>>(R1, 32, R1, 32, wpk_ec2, ec2_b, EY, 2);
  conv_wmma<9, 1, 1><<<CG, 256, 0, stream>>>(R1, 32, R1, 32, wpk_ec3, ec3_b, R2, 3);
  // softmax gate blend
  gate_mix<<<EW, 256, 0, stream>>>(EX, EY, R2, gate_w, gate_b, R3);
  // CBAM channel attention
  pool_bc<<<256, 256, 0, stream>>>(R3, AVG, MAXP);
  ca_fc<<<1, 256, 0, stream>>>(AVG, MAXP, ca1_w, ca2_w, CA);
  apply_ca<<<EW, 256, 0, stream>>>(R3, CA, R2, SAIN);
  // CBAM spatial attention + residual merge -> out0 (R3)
  sa_merge<<<EW, 256, 0, stream>>>(SAIN, sa_w, R2, R1, R3);
  // Cascaded dense stages
  conv_wmma<1, 1, 0><<<CG, 256, 0, stream>>>(R3, 32, R3, 32, wpk_p1, p1_b, ABUF, 1);
  conv_wmma<9, 1, 0><<<CG, 256, 0, stream>>>(ABUF, 32, ABUF, 32, wpk_f1, f1_b, TBUF, 1);
  conv_wmma<1, 2, 0><<<CG, 256, 0, stream>>>(R3, 32, TBUF, 32, wpk_p2, p2_b, ABUF, 1);
  conv_wmma<9, 1, 0><<<CG, 256, 0, stream>>>(ABUF, 32, ABUF, 32, wpk_f2, f2_b, R2, 1);
  conv_wmma<1, 2, 0><<<CG, 256, 0, stream>>>(R3, 32, R2, 32, wpk_p3, p3_b, ABUF, 1);
  conv_wmma<9, 1, 0><<<CG, 256, 0, stream>>>(ABUF, 32, ABUF, 32, wpk_f3, f3_b, R1, 1);
  // concat(out0, t3) -> NCHW fp32
  final_out<<<131072, 256, 0, stream>>>(R3, R1, (float*)d_out);
}
